// BahdanauAttention_81535659147981
// MI455X (gfx1250) — matrix operand order repaired and checked by reading, not compiled, not run on hardware
//
#include <hip/hip_runtime.h>
#include <stdint.h>

// ---- problem constants ----
#define BATCH   64
#define SLEN    2048
#define EDIM    1024
#define ADIM    512
#define MROWS   (BATCH * SLEN)      // 131072 GEMM rows
#define MBLK    128                 // rows per workgroup in the WMMA kernel (8 waves x 16)
#define NCHUNKS (ADIM / 16)         // 32 n-chunks
#define NEG_BIG (-3.402823466e38f)  // jnp.finfo(float32).min

typedef __bf16 v16bf __attribute__((ext_vector_type(16)));
typedef float  v8f   __attribute__((ext_vector_type(8)));

union FragU {
    v16bf v;
    uint4 q[2];
};

__device__ __forceinline__ unsigned short f2bf(float f) {
    union { float f; unsigned int u; } x; x.f = f;
    unsigned int u = x.u;
    u += 0x7fffu + ((u >> 16) & 1u);   // round-to-nearest-even
    return (unsigned short)(u >> 16);
}

// gfx1250 hardware tanh (TRANS op). ISA requires 1 independent op / V_NOP
// after a TRANS op before its result is consumed -> provide it in the asm.
__device__ __forceinline__ float htanh(float x) {
    float r;
    asm volatile("v_tanh_f32 %0, %1\n\tv_nop" : "=v"(r) : "v"(x));
    return r;
}

// ---------------------------------------------------------------------------
// Kernel 0: one-time W_enc fp32 -> bf16 conversion into workspace (1 MB).
// ---------------------------------------------------------------------------
__global__ __launch_bounds__(256) void wenc2bf_kernel(const float* __restrict__ Wenc,
                                                      unsigned short* __restrict__ Wbf) {
    int i = blockIdx.x * 256 + threadIdx.x;        // 0 .. 131071 (float4 index)
    float4 v = ((const float4*)Wenc)[i];
    unsigned int lo = (unsigned int)f2bf(v.x) | ((unsigned int)f2bf(v.y) << 16);
    unsigned int hi = (unsigned int)f2bf(v.z) | ((unsigned int)f2bf(v.w) << 16);
    ((uint2*)Wbf)[i] = make_uint2(lo, hi);
}

// ---------------------------------------------------------------------------
// Kernel 1: dec_feat[b,a] = sum_e decoder_hidden[b,e] * W_dec[a,e]   [64x512]
// ---------------------------------------------------------------------------
__global__ __launch_bounds__(256) void decfeat_kernel(const float* __restrict__ dh,
                                                      const float* __restrict__ Wdec,
                                                      float* __restrict__ decf) {
    int idx = blockIdx.x * 256 + threadIdx.x;      // 0 .. 32767
    int b = idx >> 9;
    int a = idx & 511;
    const float4* h = (const float4*)(dh + b * EDIM);
    const float4* w = (const float4*)(Wdec + a * EDIM);
    float acc = 0.f;
#pragma unroll 8
    for (int i = 0; i < EDIM / 4; ++i) {
        float4 x = h[i], y = w[i];
        acc += x.x * y.x + x.y * y.y + x.z * y.z + x.w * y.w;
    }
    decf[idx] = acc;
}

// ---------------------------------------------------------------------------
// Kernel 2: fused  scores[m] = sum_a w_e[a]*tanh( enc[m,:]·W_enc[a,:] + dec[b,a] )
// WMMA bf16 GEMM, M-tile = 128 rows (8 waves x 16). Per 16-wide n-chunk: full
// K=1024 WMMA accumulation, hardware-tanh/dot epilogue, cross-lane reduce.
// Bs staging is software-pipelined through registers (prefetch next chunk of
// pre-converted bf16 W_enc during current chunk's WMMAs).
// LDS: 256 KB A-tile (bf16) + 32 KB W_enc chunk (bf16) = 288 KB (<320 KB/WGP).
// ---------------------------------------------------------------------------
__global__ __launch_bounds__(256) void scores_kernel(const float* __restrict__ enc,
                                                     const unsigned short* __restrict__ Wbf,
                                                     const float* __restrict__ decf,
                                                     const float* __restrict__ wE,
                                                     float* __restrict__ scores) {
    __shared__ __align__(16) unsigned short As[MBLK * EDIM];   // 256 KB bf16
    __shared__ __align__(16) unsigned short Bs[16 * EDIM];     //  32 KB bf16

    const int tid  = threadIdx.x;
    const int wave = tid >> 5;
    const int lane = tid & 31;
    const int half = lane >> 4;       // 0: lanes 0-15, 1: lanes 16-31
    const int col  = lane & 15;
    const int wm   = wave * 16;       // this wave's local row base (0..112)

    const long m0 = (long)blockIdx.x * MBLK;
    const int  b  = (int)(m0 >> 11);  // MBLK divides SLEN -> one batch per WG

    // ---- stage A tile (128 rows x 1024) to LDS as bf16, float4-coalesced ----
    const float4* A4 = (const float4*)(enc + m0 * EDIM);
    for (int i = tid; i < MBLK * EDIM / 4; i += 256) {
        float4 v = A4[i];
        int o = i * 4;
        As[o + 0] = f2bf(v.x);
        As[o + 1] = f2bf(v.y);
        As[o + 2] = f2bf(v.z);
        As[o + 3] = f2bf(v.w);
    }

    const unsigned short* Arow = &As[(wm + col) * EDIM];  // A: M = lane&15 (both halves)
    const unsigned short* Brow = &Bs[col * EDIM];         // B: N = lane&15

    // ---- register prefetch of chunk 0 of bf16 W_enc (32 KB / 256 thr = 8 uint4)
    const uint4* Wq = (const uint4*)Wbf;   // 16 rows x 1024 bf16 = 2048 uint4 per chunk
    uint4* Bs4 = (uint4*)Bs;
    uint4 pref[8];
#pragma unroll
    for (int t = 0; t < 8; ++t) pref[t] = Wq[tid + t * 256];

    float sc[8];
#pragma unroll
    for (int j = 0; j < 8; ++j) sc[j] = 0.f;

    for (int nc = 0; nc < NCHUNKS; ++nc) {
        const int n0 = nc * 16;
        __syncthreads();   // previous Bs fully consumed (also covers As for nc=0)

        // ---- short LDS store burst from prefetch registers ----
#pragma unroll
        for (int t = 0; t < 8; ++t) Bs4[tid + t * 256] = pref[t];
        __syncthreads();

        // ---- issue next chunk's global loads now; consumed next iteration ----
        if (nc + 1 < NCHUNKS) {
            const uint4* src = Wq + (nc + 1) * 2048;
#pragma unroll
            for (int t = 0; t < 8; ++t) pref[t] = src[tid + t * 256];
        }

        // hoisted epilogue scalars for this n-chunk (per-lane)
        const int   n    = n0 + col;
        const float decv = decf[b * ADIM + n];
        const float wev  = wE[n];

        // ---- 16x16 tile, full K = 1024 in steps of 32 ----
        v8f c = {0.f, 0.f, 0.f, 0.f, 0.f, 0.f, 0.f, 0.f};
#pragma unroll 4
        for (int k0 = 0; k0 < EDIM; k0 += 32) {
            FragU a, bb;
            // A 16x32 bf16 layout: e0..7 -> K = k0+half*8+e ; e8..15 -> K = k0+16+half*8+(e-8)
            a.q[0] = *(const uint4*)(Arow + k0 + half * 8);
            a.q[1] = *(const uint4*)(Arow + k0 + 16 + half * 8);
            // B 32x16 bf16 layout: e0..7 -> K = k0+half*8+e ; e8..15 -> K = k0+16+half*8+(e-8) ; N = lane&15
            bb.q[0] = *(const uint4*)(Brow + k0 + half * 8);
            bb.q[1] = *(const uint4*)(Brow + k0 + 16 + half * 8);
            c = __builtin_amdgcn_wmma_f32_16x16x32_bf16(
                    false, a.v, false, bb.v, (short)0, c, false, false);
        }

        // ---- fused epilogue: + dec_feat, hardware tanh, dot with W_energy ----
#pragma unroll
        for (int j = 0; j < 8; ++j)
            sc[j] += wev * htanh(c[j] + decv);
    }

    // ---- reduce over the 16 n-columns (lanes within each half) ----
#pragma unroll
    for (int m = 1; m < 16; m <<= 1) {
#pragma unroll
        for (int j = 0; j < 8; ++j)
            sc[j] += __shfl_xor(sc[j], m, 32);
    }

    // C/D layout: row = wm + half*8 + j ; lane 0 / lane 16 hold the row sums
    if (col == 0) {
#pragma unroll
        for (int j = 0; j < 8; ++j) {
            long m = m0 + wm + half * 8 + j;
            scores[m] = sc[j];
        }
    }
}

// ---------------------------------------------------------------------------
// Kernel 3: masked softmax over S per batch row; writes attention weights.
// ---------------------------------------------------------------------------
__global__ __launch_bounds__(256) void softmax_kernel(const float* __restrict__ scores,
                                                      const uint8_t* __restrict__ mask,
                                                      float* __restrict__ attn) {
    const int b   = blockIdx.x;
    const int tid = threadIdx.x;
    __shared__ float red[8];
    __shared__ float s_max, s_sum;

    const float*   row  = scores + b * SLEN;
    const uint8_t* mrow = mask + b * SLEN;

    float mx = NEG_BIG;
    for (int s = tid; s < SLEN; s += 256) {
        float v = mrow[s] ? NEG_BIG : row[s];
        mx = fmaxf(mx, v);
    }
#pragma unroll
    for (int m = 16; m >= 1; m >>= 1) mx = fmaxf(mx, __shfl_xor(mx, m, 32));
    if ((tid & 31) == 0) red[tid >> 5] = mx;
    __syncthreads();
    if (tid == 0) {
        float m2 = red[0];
        for (int i = 1; i < 8; ++i) m2 = fmaxf(m2, red[i]);
        s_max = m2;
    }
    __syncthreads();
    mx = s_max;

    float sum = 0.f;
    for (int s = tid; s < SLEN; s += 256) {
        float v = mrow[s] ? NEG_BIG : row[s];
        sum += __expf(v - mx);
    }
#pragma unroll
    for (int m = 16; m >= 1; m >>= 1) sum += __shfl_xor(sum, m, 32);
    if ((tid & 31) == 0) red[tid >> 5] = sum;
    __syncthreads();
    if (tid == 0) {
        float t = 0.f;
        for (int i = 0; i < 8; ++i) t += red[i];
        s_sum = t;
    }
    __syncthreads();
    const float inv = 1.0f / s_sum;

    for (int s = tid; s < SLEN; s += 256) {
        float v = mrow[s] ? NEG_BIG : row[s];
        attn[b * SLEN + s] = __expf(v - mx) * inv;
    }
}

// ---------------------------------------------------------------------------
// Kernel 4: context[b,e] = sum_s attn[b,s] * enc[b,s,e]  (bandwidth bound)
// 256 WGs = 64 batches x 4 e-chunks of 256; one fp32 per thread, coalesced.
// ---------------------------------------------------------------------------
__global__ __launch_bounds__(256) void context_kernel(const float* __restrict__ enc,
                                                      const float* __restrict__ attn,
                                                      float* __restrict__ ctx) {
    const int blk = blockIdx.x;
    const int b   = blk >> 2;
    const int e   = ((blk & 3) << 8) + threadIdx.x;
    const int tid = threadIdx.x;
    __shared__ float wsh[SLEN];   // 8 KB

    for (int s = tid; s < SLEN; s += 256) wsh[s] = attn[b * SLEN + s];
    __syncthreads();

    const float* E = enc + (long)b * SLEN * EDIM + e;
    float acc0 = 0.f, acc1 = 0.f, acc2 = 0.f, acc3 = 0.f;
#pragma unroll 2
    for (int s = 0; s < SLEN; s += 4) {
        acc0 += wsh[s + 0] * E[(long)(s + 0) * EDIM];
        acc1 += wsh[s + 1] * E[(long)(s + 1) * EDIM];
        acc2 += wsh[s + 2] * E[(long)(s + 2) * EDIM];
        acc3 += wsh[s + 3] * E[(long)(s + 3) * EDIM];
    }
    ctx[b * EDIM + e] = (acc0 + acc1) + (acc2 + acc3);
}

// ---------------------------------------------------------------------------
extern "C" void kernel_launch(void* const* d_in, const int* in_sizes, int n_in,
                              void* d_out, int out_size, void* d_ws, size_t ws_size,
                              hipStream_t stream) {
    const float*   dh    = (const float*)d_in[0];   // [64,1024]
    const float*   enc   = (const float*)d_in[1];   // [64,2048,1024]
    const uint8_t* mask  = (const uint8_t*)d_in[2]; // [64,2048] bool (1 byte)
    const float*   Wenc  = (const float*)d_in[3];   // [512,1024]
    const float*   Wdec  = (const float*)d_in[4];   // [512,1024]
    const float*   Wener = (const float*)d_in[5];   // [1,512]

    float* ctx_out  = (float*)d_out;                       // [64,1024]
    float* attn_out = (float*)d_out + BATCH * EDIM;        // [64,2048]

    float*          decf      = (float*)d_ws;                          // [64,512]
    float*          scoresBuf = decf + BATCH * ADIM;                   // [64,2048]
    unsigned short* Wbf       = (unsigned short*)(scoresBuf + MROWS);  // [512,1024] bf16

    wenc2bf_kernel<<<(ADIM * EDIM / 4) / 256, 256, 0, stream>>>(Wenc, Wbf);
    decfeat_kernel<<<(BATCH * ADIM) / 256, 256, 0, stream>>>(dh, Wdec, decf);
    scores_kernel<<<MROWS / MBLK, 256, 0, stream>>>(enc, Wbf, decf, Wener, scoresBuf);
    softmax_kernel<<<BATCH, 256, 0, stream>>>(scoresBuf, mask, attn_out);
    context_kernel<<<BATCH * 4, 256, 0, stream>>>(enc, attn_out, ctx_out);
}
